// DefectGraphEncoder_38319698215655
// MI455X (gfx1250) — matrix operand order repaired and checked by reading, not compiled, not run on hardware
//
#include <hip/hip_runtime.h>
#include <hip/hip_bf16.h>
#include <cstdint>

#define NB 2
#define NN 512
#define NH 256
#define NL 2
#define NTYPES 3
#define OFFD 192
#define OUTD 128

#define MBLK 128
#define NBLK 64

typedef __attribute__((ext_vector_type(16))) _Float16 v16h;
typedef __attribute__((ext_vector_type(2)))  _Float16 h2;
typedef __attribute__((ext_vector_type(8)))  float    v8f;

__device__ __forceinline__ float silu_f(float x) {
    return x / (1.0f + __expf(-x));
}

// ---------------------------------------------------------------------------
// WMMA GEMM with async-LDS staged B operand.
//   C[M,N] = alpha * A[M,K](f16,row-major) @ Bt[N,K](f16, [n][k]) + bias[N]
// Block: 8 waves -> 128x64 macro tile; wave w owns rows [w*16, w*16+16) x 64.
// B chunk (64 rows x 32 K) double-buffered in LDS via
// global_load_async_to_lds_b128 (ASYNCcnt) shared by all 8 waves.
// Output modes: 0 = f32, 1 = f16, 2 = silu->f16, 3 = f16 transposed [b][n][m%NN]
// ---------------------------------------------------------------------------
__global__ __launch_bounds__(256) void gemm_f16_wmma(
    const _Float16* __restrict__ A, const _Float16* __restrict__ Bt,
    const float* __restrict__ bias, float* __restrict__ Cf,
    _Float16* __restrict__ Ch,
    int M, int N, int K, long sA, long sB, long sC, float alpha, int mode)
{
    __shared__ _Float16 sBuf[2][NBLK * 32];
    int tid  = threadIdx.x;
    int lane = tid & 31;
    int wave = tid >> 5;
    int gn = N / NBLK;
    int bm = blockIdx.x / gn;
    int bn = blockIdx.x - bm * gn;

    const _Float16* Ab = A + (long)blockIdx.z * sA;
    const _Float16* Bb = Bt + (long)blockIdx.z * sB;

    int m0 = bm * MBLK + wave * 16;
    int n0 = bn * NBLK;

    // B-staging coordinates: 256 lanes x 16B cover 64 rows x 32 halfs
    int lrow = tid >> 2;            // 0..63
    int lk4  = (tid & 3) * 8;       // half offset within 32-K chunk
    const _Float16* gB = Bb + (long)(n0 + lrow) * K + lk4;
    unsigned ldsB0 = (unsigned)(uintptr_t)&sBuf[0][lrow * 32 + lk4];
    unsigned ldsB1 = (unsigned)(uintptr_t)&sBuf[1][lrow * 32 + lk4];

    int am    = m0 + (lane & 15);
    int akoff = (lane >> 4) << 2;   // in h2 units: 0 or 4
    int bkoff = (lane >> 4) << 3;   // in h2 units: 0 or 8
    int bnl   = lane & 15;

    v8f acc[4] = {v8f{}, v8f{}, v8f{}, v8f{}};

    int S = K >> 5;
    asm volatile("global_load_async_to_lds_b128 %0, %1, off"
                 :: "v"(ldsB0), "v"(gB) : "memory");
    for (int k = 0; k < S; ++k) {
        if (k + 1 < S) {
            const _Float16* gBn = gB + (k + 1) * 32;
            unsigned dst = ((k + 1) & 1) ? ldsB1 : ldsB0;
            asm volatile("global_load_async_to_lds_b128 %0, %1, off"
                         :: "v"(dst), "v"(gBn) : "memory");
            asm volatile("s_wait_asynccnt 0x1" ::: "memory");
        } else {
            asm volatile("s_wait_asynccnt 0x0" ::: "memory");
        }
        __syncthreads();

        // A fragment (direct from global, L2-resident)
        v16h a;
        const h2* ap2 = (const h2*)(Ab + (long)am * K + k * 32);
#pragma unroll
        for (int r = 0; r < 4; ++r) {
            h2 p0 = ap2[akoff + r];
            h2 p1 = ap2[8 + akoff + r];
            a[2 * r]     = p0[0];
            a[2 * r + 1] = p0[1];
            a[2 * r + 8] = p1[0];
            a[2 * r + 9] = p1[1];
        }
        // 4 B fragments from LDS, 4 WMMAs
        const h2* sb2 = (const h2*)&sBuf[k & 1][0];
#pragma unroll
        for (int nt = 0; nt < 4; ++nt) {
            int rowb = (nt * 16 + bnl) * 16;
            v16h b;
#pragma unroll
            for (int r = 0; r < 8; ++r) {
                h2 pp = sb2[rowb + akoff + (r & 3) + ((r >> 2) << 3)];
                b[2 * r]     = pp[0];
                b[2 * r + 1] = pp[1];
            }
            acc[nt] = __builtin_amdgcn_wmma_f32_16x16x32_f16(
                false, a, false, b, (short)0, acc[nt], false, false);
        }
        __syncthreads();
    }

    // epilogue
    int mbase = m0 + ((lane >> 4) << 3);
#pragma unroll
    for (int nt = 0; nt < 4; ++nt) {
        int n = n0 + nt * 16 + bnl;
        float bv = bias ? bias[n] : 0.0f;
#pragma unroll
        for (int r = 0; r < 8; ++r) {
            int m = mbase + r;
            float y = acc[nt][r] * alpha + bv;
            if (mode == 0) {
                Cf[(long)blockIdx.z * sC + (long)m * N + n] = y;
            } else if (mode == 1) {
                Ch[(long)blockIdx.z * sC + (long)m * N + n] = (_Float16)y;
            } else if (mode == 2) {
                Ch[(long)blockIdx.z * sC + (long)m * N + n] = (_Float16)silu_f(y);
            } else {
                // transposed f16: [b][n][m%NN] (v -> vT for attn@v)
                Ch[((long)(m >> 9) * N + n) * NN + (m & (NN - 1))] = (_Float16)y;
            }
        }
    }
}

// W[K,N] fp32 row-major -> Wt[N,K] f16
__global__ void wtrans_kernel(const float* __restrict__ W,
                              _Float16* __restrict__ Wt, int K, int N)
{
    long i = (long)blockIdx.x * 256 + threadIdx.x;
    if (i >= (long)K * N) return;
    int n = (int)(i / K);
    int k = (int)(i - (long)n * K);
    Wt[i] = (_Float16)W[(long)k * N + n];
}

// Node embedding: off MLP (3->192 silu ->192) + type embedding concat.
__global__ __launch_bounds__(256) void embed_kernel(
    const float* __restrict__ node_attr, const float* __restrict__ type_emb,
    const float* __restrict__ ow1, const float* __restrict__ ob1,
    const float* __restrict__ ow2, const float* __restrict__ ob2,
    float* __restrict__ offsets, _Float16* __restrict__ hpre16)
{
    int node = blockIdx.x;
    int t = threadIdx.x;
    __shared__ float u[OFFD];
    __shared__ int stype;
    float a0 = node_attr[node * 4 + 0] * 0.125f;
    float a1 = node_attr[node * 4 + 1] * 0.125f;
    float a2 = node_attr[node * 4 + 2] * 0.125f;
    if (t < 3) offsets[node * 3 + t] = node_attr[node * 4 + t] * 0.125f;
    if (t == 0) {
        int ty = (int)node_attr[node * 4 + 3];
        ty = ty < 0 ? 0 : (ty > NTYPES - 1 ? NTYPES - 1 : ty);
        stype = ty;
    }
    if (t < OFFD) {
        float s = a0 * ow1[0 * OFFD + t] + a1 * ow1[1 * OFFD + t] +
                  a2 * ow1[2 * OFFD + t] + ob1[t];
        u[t] = silu_f(s);
    }
    __syncthreads();
    float y;
    if (t < OFFD) {
        y = ob2[t];
        for (int pI = 0; pI < OFFD; ++pI) y += u[pI] * ow2[pI * OFFD + t];
    } else {
        y = type_emb[stype * 64 + (t - OFFD)];
    }
    hpre16[(long)node * NH + t] = (_Float16)y;
}

// LayerNorm (+optional residual, silu, node-mask); also emits f16 copy.
__global__ __launch_bounds__(256) void ln_kernel(
    float* __restrict__ h, const float* __restrict__ delta,
    const float* __restrict__ g, const float* __restrict__ b,
    const float* __restrict__ mask, int flags, _Float16* __restrict__ out16)
{
    int node = blockIdx.x;
    int t = threadIdx.x;
    __shared__ float s1[256], s2[256];
    float x = h[(long)node * NH + t];
    if (delta) x += delta[(long)node * NH + t];
    s1[t] = x;
    s2[t] = x * x;
    __syncthreads();
    for (int s = 128; s > 0; s >>= 1) {
        if (t < s) { s1[t] += s1[t + s]; s2[t] += s2[t + s]; }
        __syncthreads();
    }
    float mean = s1[0] * (1.0f / NH);
    float var  = s2[0] * (1.0f / NH) - mean * mean;
    float y = (x - mean) * rsqrtf(var + 1e-5f) * g[t] + b[t];
    if (flags & 1) y = silu_f(y);
    if (flags & 2) y *= mask[node];
    h[(long)node * NH + t] = y;
    if (out16) out16[(long)node * NH + t] = (_Float16)y;
}

// Per-pair edge-offset MLP bias for BOTH layers (grid.z = layer) + pair mask.
__global__ __launch_bounds__(256) void edge_kernel(
    const float* __restrict__ offs, const float* __restrict__ nmask,
    const float* __restrict__ ew1, const float* __restrict__ eb1,
    const float* __restrict__ ew2, const float* __restrict__ eb2,
    float* __restrict__ biasbuf, float* __restrict__ pmask)
{
    int l = blockIdx.z;
    int tid = threadIdx.x;
    __shared__ float sw1[3 * NH], sb1[NH], sw2[NH];
    sw1[tid]          = ew1[(l * 3 + 0) * NH + tid];
    sw1[NH + tid]     = ew1[(l * 3 + 1) * NH + tid];
    sw1[2 * NH + tid] = ew1[(l * 3 + 2) * NH + tid];
    sb1[tid] = eb1[l * NH + tid];
    sw2[tid] = ew2[l * NH + tid];
    __syncthreads();
    long idx = (long)blockIdx.x * 256 + tid;   // < B*N*N
    int bb  = (int)(idx >> 18);                // N*N = 2^18
    int rem = (int)(idx & 262143);
    int i = rem >> 9, j = rem & 511;
    const float* oi = offs + ((long)bb * NN + i) * 3;
    const float* oj = offs + ((long)bb * NN + j) * 3;
    float dx = oi[0] - oj[0], dy = oi[1] - oj[1], dz = oi[2] - oj[2];
    float acc = eb2[l];
    for (int hh = 0; hh < NH; ++hh) {
        float s = dx * sw1[hh] + dy * sw1[NH + hh] + dz * sw1[2 * NH + hh] + sb1[hh];
        acc += silu_f(s) * sw2[hh];
    }
    biasbuf[(long)l * NB * NN * NN + idx] = acc;
    if (l == 0) {
        float d2 = dx * dx + dy * dy + dz * dz;
        float pm = nmask[bb * NN + i] * nmask[bb * NN + j] *
                   (d2 <= 4.000001f ? 1.0f : 0.0f);
        pmask[idx] = pm;
    }
}

// Masked softmax + renormalize; writes f16 attention.
__global__ __launch_bounds__(256) void softmax_kernel(
    const float* __restrict__ scores, const float* __restrict__ biasl,
    const float* __restrict__ pmask, _Float16* __restrict__ attn16)
{
    int row = blockIdx.x;    // b*N + i
    int t = threadIdx.x;
    __shared__ float red[256];
    long base = (long)row * NN;
    float s0 = scores[base + t]       + biasl[base + t];
    float s1 = scores[base + t + 256] + biasl[base + t + 256];
    float m0 = pmask[base + t], m1 = pmask[base + t + 256];
    float x0 = (m0 > 0.f) ? s0 : -1e30f;
    float x1 = (m1 > 0.f) ? s1 : -1e30f;
    red[t] = fmaxf(x0, x1);
    __syncthreads();
    for (int s = 128; s > 0; s >>= 1) {
        if (t < s) red[t] = fmaxf(red[t], red[t + s]);
        __syncthreads();
    }
    float rowmax = red[0];
    __syncthreads();
    float e0 = (m0 > 0.f) ? __expf(x0 - rowmax) : 0.f;
    float e1 = (m1 > 0.f) ? __expf(x1 - rowmax) : 0.f;
    red[t] = e0 + e1;
    __syncthreads();
    for (int s = 128; s > 0; s >>= 1) {
        if (t < s) red[t] += red[t + s];
        __syncthreads();
    }
    float denom = red[0];
    float inv = (denom > 0.f) ? 1.0f / denom : 0.f;
    attn16[base + t]       = (_Float16)(e0 * inv);
    attn16[base + t + 256] = (_Float16)(e1 * inv);
}

// Masked mean pool + output head. One block per batch.
__global__ __launch_bounds__(256) void pool_out_kernel(
    const float* __restrict__ h, const float* __restrict__ nmask,
    const float* __restrict__ ow, const float* __restrict__ ob,
    float* __restrict__ out)
{
    int b = blockIdx.x;
    int t = threadIdx.x;
    __shared__ float pooled[NH];
    __shared__ float red[256];
    float acc = 0.f;
    for (int n = 0; n < NN; ++n) acc += h[((long)b * NN + n) * NH + t];
    float ms = 0.f;
    for (int n = t; n < NN; n += 256) ms += nmask[b * NN + n];
    red[t] = ms;
    __syncthreads();
    for (int s = 128; s > 0; s >>= 1) {
        if (t < s) red[t] += red[t + s];
        __syncthreads();
    }
    float den = fmaxf(red[0], 1.0f);
    pooled[t] = acc / den;
    __syncthreads();
    if (t < OUTD) {
        float o = ob[t];
        for (int hh = 0; hh < NH; ++hh) o += pooled[hh] * ow[hh * OUTD + t];
        out[b * OUTD + t] = o;
    }
}

extern "C" void kernel_launch(void* const* d_in, const int* in_sizes, int n_in,
                              void* d_out, int out_size, void* d_ws, size_t ws_size,
                              hipStream_t stream) {
    const float* node_attr = (const float*)d_in[0];
    const float* node_mask = (const float*)d_in[1];
    const float* type_emb  = (const float*)d_in[2];
    const float* off_w1 = (const float*)d_in[3];
    const float* off_b1 = (const float*)d_in[4];
    const float* off_w2 = (const float*)d_in[5];
    const float* off_b2 = (const float*)d_in[6];
    const float* in_w  = (const float*)d_in[7];
    const float* in_b  = (const float*)d_in[8];
    const float* in_g  = (const float*)d_in[9];
    const float* in_bn = (const float*)d_in[10];
    const float* q_w = (const float*)d_in[11];
    const float* q_b = (const float*)d_in[12];
    const float* k_w = (const float*)d_in[13];
    const float* k_b = (const float*)d_in[14];
    const float* v_w = (const float*)d_in[15];
    const float* v_b = (const float*)d_in[16];
    const float* o_w = (const float*)d_in[17];
    const float* o_b = (const float*)d_in[18];
    const float* e_w1 = (const float*)d_in[19];
    const float* e_b1 = (const float*)d_in[20];
    const float* e_w2 = (const float*)d_in[21];
    const float* e_b2 = (const float*)d_in[22];
    const float* ln1_g = (const float*)d_in[23];
    const float* ln1_b = (const float*)d_in[24];
    const float* ff_w1 = (const float*)d_in[25];
    const float* ff_b1 = (const float*)d_in[26];
    const float* ff_w2 = (const float*)d_in[27];
    const float* ff_b2 = (const float*)d_in[28];
    const float* ln2_g = (const float*)d_in[29];
    const float* ln2_b = (const float*)d_in[30];
    const float* out_w = (const float*)d_in[31];
    const float* out_b = (const float*)d_in[32];
    float* out = (float*)d_out;

    // ---- workspace layout ----
    char* p = (char*)d_ws;
    auto alloc = [&](size_t elems, size_t esz) {
        void* r = (void*)p;
        p += ((elems * esz) + 255) & ~(size_t)255;
        return r;
    };
    const long HB = (long)NB * NN * NH;       // 262144
    const long SB = (long)NB * NN * NN;       // 524288
    _Float16* inT   = (_Float16*)alloc(NH * NH, 2);
    _Float16* qT    = (_Float16*)alloc((size_t)NL * NH * NH, 2);
    _Float16* kT    = (_Float16*)alloc((size_t)NL * NH * NH, 2);
    _Float16* vT    = (_Float16*)alloc((size_t)NL * NH * NH, 2);
    _Float16* oT    = (_Float16*)alloc((size_t)NL * NH * NH, 2);
    _Float16* ff1T  = (_Float16*)alloc((size_t)NL * NH * 2 * NH, 2);
    _Float16* ff2T  = (_Float16*)alloc((size_t)NL * 2 * NH * NH, 2);
    _Float16* hpre16 = (_Float16*)alloc(HB, 2);
    _Float16* h16    = (_Float16*)alloc(HB, 2);
    _Float16* q16    = (_Float16*)alloc(HB, 2);
    _Float16* k16    = (_Float16*)alloc(HB, 2);
    _Float16* vt16   = (_Float16*)alloc(HB, 2);
    _Float16* attn16 = (_Float16*)alloc(SB, 2);
    _Float16* mid16  = (_Float16*)alloc((size_t)NB * NN * 2 * NH, 2);
    _Float16* upd16  = (_Float16*)alloc(HB, 2);
    float* offsets = (float*)alloc((size_t)NB * NN * 3, 4);
    float* h       = (float*)alloc(HB, 4);
    float* qf      = (float*)alloc(HB, 4);
    float* updf    = (float*)alloc(HB, 4);
    float* scores  = (float*)alloc(SB, 4);
    float* biasbuf = (float*)alloc((size_t)NL * SB, 4);
    float* pmask   = (float*)alloc(SB, 4);

    auto gemm = [&](const _Float16* A, const _Float16* Bt, const float* bias,
                    float* Cf, _Float16* Ch, int M, int N, int K,
                    long sA, long sB2, long sC, int batch, float alpha, int mode) {
        dim3 grid((unsigned)((M / MBLK) * (N / NBLK)), 1, batch);
        gemm_f16_wmma<<<grid, 256, 0, stream>>>(A, Bt, bias, Cf, Ch, M, N, K,
                                                sA, sB2, sC, alpha, mode);
    };
    auto wt = [&](const float* W, _Float16* Wt, int K, int N) {
        long n = (long)K * N;
        wtrans_kernel<<<(unsigned)((n + 255) / 256), 256, 0, stream>>>(W, Wt, K, N);
    };

    // 1) weight prep: fp32 [K,N] -> f16 [N,K]
    wt(in_w, inT, NH, NH);
    for (int l = 0; l < NL; ++l) {
        wt(q_w + (long)l * NH * NH, qT + (long)l * NH * NH, NH, NH);
        wt(k_w + (long)l * NH * NH, kT + (long)l * NH * NH, NH, NH);
        wt(v_w + (long)l * NH * NH, vT + (long)l * NH * NH, NH, NH);
        wt(o_w + (long)l * NH * NH, oT + (long)l * NH * NH, NH, NH);
        wt(ff_w1 + (long)l * NH * 2 * NH, ff1T + (long)l * NH * 2 * NH, NH, 2 * NH);
        wt(ff_w2 + (long)l * 2 * NH * NH, ff2T + (long)l * 2 * NH * NH, 2 * NH, NH);
    }

    // 2) node embedding; in_w GEMM; LN + SiLU + mask (also emits h16)
    embed_kernel<<<NB * NN, 256, 0, stream>>>(node_attr, type_emb, off_w1,
                                              off_b1, off_w2, off_b2,
                                              offsets, hpre16);
    gemm(hpre16, inT, in_b, h, nullptr, NB * NN, NH, NH, 0, 0, 0, 1, 1.0f, 0);
    ln_kernel<<<NB * NN, 256, 0, stream>>>(h, nullptr, in_g, in_bn, node_mask,
                                           3, h16);

    // 3) edge biases (both layers) + pair mask
    {
        dim3 eg((unsigned)(SB / 256), 1, NL);
        edge_kernel<<<eg, 256, 0, stream>>>(offsets, node_mask, e_w1, e_b1,
                                            e_w2, e_b2, biasbuf, pmask);
    }

    const float inv_sqrt_h = 0.0625f;   // 1/sqrt(256)
    for (int l = 0; l < NL; ++l) {
        long wo = (long)l * NH * NH;
        // q,k -> f16 direct; v -> transposed f16 direct
        gemm(h16, qT + wo, q_b + l * NH, nullptr, q16, NB * NN, NH, NH,
             0, 0, (long)NB * NN * NH, 1, 1.0f, 1);
        gemm(h16, kT + wo, k_b + l * NH, nullptr, k16, NB * NN, NH, NH,
             0, 0, (long)NB * NN * NH, 1, 1.0f, 1);
        gemm(h16, vT + wo, v_b + l * NH, nullptr, vt16, NB * NN, NH, NH,
             0, 0, 0, 1, 1.0f, 3);
        // scores = q @ k^T / sqrt(H)
        gemm(q16, k16, nullptr, scores, nullptr, NN, NN, NH,
             (long)NN * NH, (long)NN * NH, (long)NN * NN, NB, inv_sqrt_h, 0);
        // fused bias + mask + softmax -> attn16
        softmax_kernel<<<NB * NN, 256, 0, stream>>>(
            scores, biasbuf + (long)l * SB, pmask, attn16);
        // upd = attn @ v  (f16 out, consumed by o-projection)
        gemm(attn16, vt16, nullptr, nullptr, upd16, NN, NH, NN,
             (long)NN * NN, (long)NH * NN, (long)NN * NH, NB, 1.0f, 1);
        // o projection (f32 out as residual delta)
        gemm(upd16, oT + wo, o_b + l * NH, qf, nullptr, NB * NN, NH, NH,
             0, 0, 0, 1, 1.0f, 0);
        // h = LN(h + o_out); emits h16 for FF
        ln_kernel<<<NB * NN, 256, 0, stream>>>(h, qf, ln1_g + l * NH,
                                               ln1_b + l * NH, nullptr, 0, h16);
        // FF: ff1 with fused SiLU->f16; ff2 f32 out
        gemm(h16, ff1T + (long)l * NH * 2 * NH, ff_b1 + l * 2 * NH,
             nullptr, mid16, NB * NN, 2 * NH, NH, 0, 0, 0, 1, 1.0f, 2);
        gemm(mid16, ff2T + (long)l * 2 * NH * NH, ff_b2 + l * NH,
             updf, nullptr, NB * NN, NH, 2 * NH, 0, 0, 0, 1, 1.0f, 0);
        // h = LN(h + ff) * mask; emits h16 for next layer
        ln_kernel<<<NB * NN, 256, 0, stream>>>(h, updf, ln2_g + l * NH,
                                               ln2_b + l * NH, node_mask, 2, h16);
    }

    // 4) masked mean pool + output head
    pool_out_kernel<<<NB, 256, 0, stream>>>(h, node_mask, out_w, out_b, out);
}
